// corner_pool_66812511257278
// MI455X (gfx1250) — matrix operand order repaired and checked by reading, not compiled, not run on hardware
//
#include <hip/hip_runtime.h>
#include <hip/hip_bf16.h>

// ---------------------------------------------------------------------------
// Vector types
// ---------------------------------------------------------------------------
typedef __attribute__((ext_vector_type(16))) __bf16         v16bf;
typedef __attribute__((ext_vector_type(8)))  float          v8f;
typedef __attribute__((ext_vector_type(8)))  unsigned short v8u16;
typedef __attribute__((ext_vector_type(4)))  unsigned int   v4u;
typedef __attribute__((ext_vector_type(8)))  int            v8i;
typedef __attribute__((ext_vector_type(4)))  int            v4i;

union Frag16 { v16bf bf; v8u16 u[2]; };

#if __has_builtin(__builtin_amdgcn_tensor_load_to_lds)
#define HAVE_TDM 1
#else
#define HAVE_TDM 0
#endif

__device__ __forceinline__ unsigned short f2bf(float f) {
    unsigned u = __float_as_uint(f);
    unsigned r = u + 0x7FFFu + ((u >> 16) & 1u);   // RNE
    return (unsigned short)(r >> 16);
}
__device__ __forceinline__ float bf2f(unsigned short h) {
    return __uint_as_float((unsigned)h << 16);
}

// fixed problem dims
#define HH 128
#define WW 128
#define OCC 256

// ---------------------------------------------------------------------------
// Weight pack: fold BN scale, bf16, layout [oc][tap][ic]; bias = beta - mean*s
// ---------------------------------------------------------------------------
__global__ void pack_w_kernel(const float* __restrict__ w,
                              const float* __restrict__ g,
                              const float* __restrict__ bt,
                              const float* __restrict__ mu,
                              const float* __restrict__ vr,
                              unsigned short* __restrict__ dst,
                              float* __restrict__ bias,
                              int OC, int IC, int taps) {
    size_t id = (size_t)blockIdx.x * blockDim.x + threadIdx.x;
    size_t total = (size_t)OC * IC * taps;
    if (id >= total) return;
    int o   = (int)(id / ((size_t)IC * taps));
    int rem = (int)(id % ((size_t)IC * taps));
    int i   = rem / taps;
    int t   = rem % taps;
    float s = g[o] * rsqrtf(vr[o] + 1e-5f);
    dst[(size_t)o * IC * taps + (size_t)t * IC + i] = f2bf(w[id] * s);
    if (rem == 0) bias[o] = bt[o] - mu[o] * s;
}

__global__ void f2bf_kernel(const float* __restrict__ x,
                            unsigned short* __restrict__ y, size_t n) {
    size_t id = (size_t)blockIdx.x * blockDim.x + threadIdx.x;
    if (id < n) y[id] = f2bf(x[id]);
}

// ---------------------------------------------------------------------------
// One conv accumulation phase.
//   Block tile: 64 oc x 128 px (one full output row). 8 waves, each 16oc x 64px.
//   K order: dy-major; per (dy, 32-ic chunk): B row tile loaded once with zero
//   halo rows and reused for dx = -1,0,1 (shifted LDS rows).
//   A tile [oc][tap][ic-chunk] loaded via TDM (tensor_load_to_lds, padded) or
//   manual vector loads.
// ---------------------------------------------------------------------------
template<int IC, int NT>
__device__ __forceinline__ void conv_phase(const unsigned short* __restrict__ inp,
                                           const unsigned short* __restrict__ wpk,
                                           unsigned short* Asm, unsigned short* Bsm,
                                           int bb, int oy, int oc0,
                                           int ocsub, int pxhalf, int lane, int tid, int wave,
                                           v8f acc[4]) {
    constexpr int NDY = (NT == 9) ? 3 : 1;
    constexpr int NDX = (NT == 9) ? 3 : 1;

    for (int dy = 0; dy < NDY; ++dy) {
        const int iy  = oy + dy - (NDY == 3 ? 1 : 0);
        const bool yok = ((unsigned)iy < (unsigned)HH);
        const int t0 = dy * NDX;

        #pragma unroll 1
        for (int ic0 = 0; ic0 < IC; ic0 += 32) {
            __syncthreads();   // previous stage fully consumed

            // ---- B tile: 32 ic x 128 px -> Bsm rows 1..128 ([px][k], stride 40)
            {
                int j   = tid >> 3;            // ic in chunk 0..31
                int p16 = (tid & 7) << 4;      // px group of 16
                unsigned short vals[16];
                if (yok) {
                    const unsigned short* src = inp
                        + (((size_t)bb * IC + (ic0 + j)) * HH + iy) * (size_t)WW;
                    *(v8u16*)&vals[0] = *(const v8u16*)(src + p16);
                    *(v8u16*)&vals[8] = *(const v8u16*)(src + p16 + 8);
                } else {
                    #pragma unroll
                    for (int e = 0; e < 16; ++e) vals[e] = 0;
                }
                #pragma unroll
                for (int e = 0; e < 16; ++e)
                    Bsm[(1 + p16 + e) * 40 + j] = vals[e];
            }

            // ---- A tile: 64 oc x NDX taps x 32 ic -> Asm row (oc*NDX+tap), stride 40
#if HAVE_TDM
            if (wave == 0) {
                const unsigned short* srcA = wpk + (size_t)oc0 * (IC * NT)
                                                 + (size_t)t0 * IC + ic0;
                unsigned long long ga = (unsigned long long)(uintptr_t)srcA;
                unsigned ldsa = (unsigned)(uintptr_t)Asm;
                v4u g0;
                g0[0] = 1u;                                     // count=1, user mode
                g0[1] = ldsa;                                   // lds_addr
                g0[2] = (unsigned)ga;                           // global_addr lo
                g0[3] = (unsigned)(ga >> 32) | (2u << 30);      // global_addr hi | type=2
                v8i g1;
                // data_size=2B(code1), pad_enable, pad_interval=16DW(code3), pad_amount=4DW(code3)
                g1[0] = (int)((1u << 16) | (1u << 20) | (3u << 22) | (3u << 25));
                g1[1] = (int)(32u << 16);                       // tensor_dim0=32 (lo16)
                g1[2] = (int)((unsigned)NDX << 16);             // dim0 hi=0 | tensor_dim1=NDX (lo16)
                g1[3] = (int)(32u << 16);                       // dim1 hi=0 | tile_dim0=32
                g1[4] = (int)((unsigned)NDX | (64u << 16));     // tile_dim1=NDX | tile_dim2=64
                unsigned long long s0 = (unsigned long long)IC;          // tap stride (elems)
                unsigned long long s1 = (unsigned long long)IC * NT;     // oc stride (elems)
                g1[5] = (int)(unsigned)s0;
                g1[6] = (int)(((unsigned)(s0 >> 32) & 0xFFFFu) | (((unsigned)s1 & 0xFFFFu) << 16));
                g1[7] = (int)(unsigned)(s1 >> 16);
                v4i g2;
                g2[0] = 64;                                     // tensor_dim2
                g2[1] = 0; g2[2] = 0; g2[3] = 0;
                v4i g3 = (v4i)0;
                v8i g4 = (v8i)0;                                // unused trailing group (6-arg form)
                __builtin_amdgcn_tensor_load_to_lds(g0, g1, g2, g3, g4, 0);
                __builtin_amdgcn_s_wait_tensorcnt(0);
            }
#else
            {
                int row = tid >> 2;                 // 0..63 (oc)
                int seg = (tid & 3) << 3;           // 0,8,16,24
                #pragma unroll
                for (int ta = 0; ta < NDX; ++ta) {
                    const unsigned short* srcA = wpk
                        + (size_t)(oc0 + row) * (IC * NT)
                        + (size_t)(t0 + ta) * IC + ic0 + seg;
                    *(v8u16*)&Asm[(row * NDX + ta) * 40 + seg] = *(const v8u16*)srcA;
                }
            }
#endif
            __syncthreads();

            // ---- compute: NDX taps x 4 px-fragments of WMMA
            #pragma unroll
            for (int dx = 0; dx < NDX; ++dx) {
                Frag16 fa;
                {
                    int arow = (ocsub + (lane & 15)) * NDX + dx;
                    int kb   = (lane >> 4) << 3;      // 0 / 8
                    fa.u[0] = *(const v8u16*)&Asm[arow * 40 + kb];
                    fa.u[1] = *(const v8u16*)&Asm[arow * 40 + kb + 16];
                }
                const int shift = (NDX == 3) ? (dx - 1) : 0;
                const int kb2   = (lane >> 4) << 3;   // 0 / 8
                #pragma unroll
                for (int g = 0; g < 4; ++g) {
                    int brow = 1 + shift + pxhalf + (g << 4) + (lane & 15);
                    Frag16 fb;
                    fb.u[0] = *(const v8u16*)&Bsm[brow * 40 + kb2];
                    fb.u[1] = *(const v8u16*)&Bsm[brow * 40 + kb2 + 16];
                    acc[g] = __builtin_amdgcn_wmma_f32_16x16x32_bf16(
                                 false, fa.bf, false, fb.bf, (short)0, acc[g], false, false);
                }
            }
        }
    }
}

template<int IC1, int NT1, int IC2, int NT2, bool OUTF32>
__global__ __launch_bounds__(256)
void conv_wmma_kernel(const unsigned short* __restrict__ in1,
                      const unsigned short* __restrict__ wp1,
                      const float* __restrict__ bias1,
                      const unsigned short* __restrict__ in2,
                      const unsigned short* __restrict__ wp2,
                      const float* __restrict__ bias2,
                      unsigned short* __restrict__ out_bf,
                      float* __restrict__ out_f32) {
    // grid = (OC/64) * H * B ; all power-of-two decode
    int bx  = blockIdx.x;
    const int oc0 = (bx & 3) << 6;
    const int oy  = (bx >> 2) & (HH - 1);
    const int bb  = bx >> 9;

    __shared__ unsigned short Asm[192 * 40];   // [oc*NDX+tap][k], 80B row stride
    __shared__ unsigned short Bsm[130 * 40];   // [1+px][k], rows 0 & 129 = zero halo

    const int tid    = threadIdx.x;
    const int lane   = tid & 31;
    const int wave   = tid >> 5;
    const int ocsub  = (wave >> 1) << 4;       // 0,16,32,48
    const int pxhalf = (wave & 1) << 6;        // 0,64

    if (tid < 40) { Bsm[tid] = 0; Bsm[129 * 40 + tid] = 0; }

    v8f acc[4] = {v8f{}, v8f{}, v8f{}, v8f{}};

    conv_phase<IC1, NT1>(in1, wp1, Asm, Bsm, bb, oy, oc0, ocsub, pxhalf, lane, tid, wave, acc);
    if constexpr (IC2 > 0) {
        conv_phase<IC2, NT2>(in2, wp2, Asm, Bsm, bb, oy, oc0, ocsub, pxhalf, lane, tid, wave, acc);
    }

    // ---- epilogue: bias (+ fused skip bias) + ReLU + store ----
    const int mbase = ocsub + ((lane >> 4) << 3);
    #pragma unroll
    for (int g = 0; g < 4; ++g) {
        int px = pxhalf + (g << 4) + (lane & 15);
        #pragma unroll
        for (int r = 0; r < 8; ++r) {
            int oc = oc0 + mbase + r;
            float v = acc[g][r] + bias1[oc];
            if constexpr (IC2 > 0) v += bias2[oc];
            v = fmaxf(v, 0.0f);
            size_t addr = (((size_t)bb * OCC + oc) * HH + oy) * (size_t)WW + px;
            if constexpr (OUTF32) out_f32[addr] = v;
            else                  out_bf[addr]  = f2bf(v);
        }
    }
}

// ---------------------------------------------------------------------------
// Reverse cummax along W on p2 (channels D..2D-1 of p12) -> tmp2
// ---------------------------------------------------------------------------
__global__ void pool_w_kernel(const unsigned short* __restrict__ p12,
                              unsigned short* __restrict__ tmp2,
                              int Bn, int D) {
    int idx = blockIdx.x * blockDim.x + threadIdx.x;   // over B*D*H
    if (idx >= Bn * D * HH) return;
    int h = idx & (HH - 1); int r = idx >> 7;
    int d = r & (D - 1);    int b = r >> 7;
    const unsigned short* src = p12 + (((size_t)b * OCC + (D + d)) * HH + h) * (size_t)WW;
    unsigned short* dst       = tmp2 + (((size_t)b * D + d) * HH + h) * (size_t)WW;
    float m = -3.4e38f;
    for (int w = WW - 1; w >= 0; --w) {
        m = fmaxf(m, bf2f(src[w]));
        dst[w] = f2bf(m);
    }
}

// ---------------------------------------------------------------------------
// Reverse cummax along H on p1 (channels 0..D-1) + tmp2 -> psum (bf16)
// ---------------------------------------------------------------------------
__global__ void pool_h_sum_kernel(const unsigned short* __restrict__ p12,
                                  const unsigned short* __restrict__ tmp2,
                                  unsigned short* __restrict__ psum,
                                  int Bn, int D) {
    int idx = blockIdx.x * blockDim.x + threadIdx.x;   // over B*D*W
    if (idx >= Bn * D * WW) return;
    int w = idx & (WW - 1); int r = idx >> 7;
    int d = r & (D - 1);    int b = r >> 7;
    const unsigned short* src = p12 + (((size_t)b * OCC + d) * HH) * (size_t)WW + w;
    const unsigned short* t2  = tmp2 + (((size_t)b * D + d) * HH) * (size_t)WW + w;
    unsigned short* dst       = psum + (((size_t)b * D + d) * HH) * (size_t)WW + w;
    float m = -3.4e38f;
    for (int h = HH - 1; h >= 0; --h) {
        m = fmaxf(m, bf2f(src[(size_t)h * WW]));
        dst[(size_t)h * WW] = f2bf(m + bf2f(t2[(size_t)h * WW]));
    }
}

// ---------------------------------------------------------------------------
// Host launcher
// ---------------------------------------------------------------------------
extern "C" void kernel_launch(void* const* d_in, const int* in_sizes, int n_in,
                              void* d_out, int out_size, void* d_ws, size_t ws_size,
                              hipStream_t stream) {
    (void)in_sizes; (void)n_in; (void)out_size; (void)ws_size;
    const int Bn = 8, C = 256, D = 128;

    const float* x    = (const float*)d_in[0];
    const float* w_p1 = (const float*)d_in[1];
    const float* g_p1 = (const float*)d_in[2];
    const float* b_p1 = (const float*)d_in[3];
    const float* m_p1 = (const float*)d_in[4];
    const float* v_p1 = (const float*)d_in[5];
    const float* w_p2 = (const float*)d_in[6];
    const float* g_p2 = (const float*)d_in[7];
    const float* b_p2 = (const float*)d_in[8];
    const float* m_p2 = (const float*)d_in[9];
    const float* v_p2 = (const float*)d_in[10];
    const float* w_p  = (const float*)d_in[11];
    const float* g_p  = (const float*)d_in[12];
    const float* b_p  = (const float*)d_in[13];
    const float* m_p  = (const float*)d_in[14];
    const float* v_p  = (const float*)d_in[15];
    const float* w_c1 = (const float*)d_in[16];
    const float* g_c1 = (const float*)d_in[17];
    const float* b_c1 = (const float*)d_in[18];
    const float* m_c1 = (const float*)d_in[19];
    const float* v_c1 = (const float*)d_in[20];
    const float* w_c2 = (const float*)d_in[21];
    const float* g_c2 = (const float*)d_in[22];
    const float* b_c2 = (const float*)d_in[23];
    const float* m_c2 = (const float*)d_in[24];
    const float* v_c2 = (const float*)d_in[25];

    char* ws = (char*)d_ws;
    size_t off = 0;
    auto take = [&](size_t bytes) -> char* {
        char* p = ws + off;
        off = (off + bytes + 255) & ~(size_t)255;
        return p;
    };
    unsigned short* w12  = (unsigned short*)take((size_t)C * C * 9 * 2);
    float*          bi12 = (float*)take((size_t)C * 4);
    unsigned short* wpP  = (unsigned short*)take((size_t)C * D * 9 * 2);
    float*          biP  = (float*)take((size_t)C * 4);
    unsigned short* wc1  = (unsigned short*)take((size_t)C * C * 2);
    float*          bic1 = (float*)take((size_t)C * 4);
    unsigned short* wc2  = (unsigned short*)take((size_t)C * C * 9 * 2);
    float*          bic2 = (float*)take((size_t)C * 4);
    unsigned short* xbf  = (unsigned short*)take((size_t)Bn * C * HH * WW * 2);
    unsigned short* p12  = (unsigned short*)take((size_t)Bn * C * HH * WW * 2);
    unsigned short* tmp2 = (unsigned short*)take((size_t)Bn * D * HH * WW * 2);
    unsigned short* psum = (unsigned short*)take((size_t)Bn * D * HH * WW * 2);
    unsigned short* rl1  = (unsigned short*)take((size_t)Bn * C * HH * WW * 2);

    const int thr = 256;

    { size_t n = (size_t)D * C * 9;
      pack_w_kernel<<<(unsigned)((n + thr - 1) / thr), thr, 0, stream>>>(
          w_p1, g_p1, b_p1, m_p1, v_p1, w12, bi12, D, C, 9);
      pack_w_kernel<<<(unsigned)((n + thr - 1) / thr), thr, 0, stream>>>(
          w_p2, g_p2, b_p2, m_p2, v_p2, w12 + (size_t)D * C * 9, bi12 + D, D, C, 9); }
    { size_t n = (size_t)C * D * 9;
      pack_w_kernel<<<(unsigned)((n + thr - 1) / thr), thr, 0, stream>>>(
          w_p, g_p, b_p, m_p, v_p, wpP, biP, C, D, 9); }
    { size_t n = (size_t)C * C;
      pack_w_kernel<<<(unsigned)((n + thr - 1) / thr), thr, 0, stream>>>(
          w_c1, g_c1, b_c1, m_c1, v_c1, wc1, bic1, C, C, 1); }
    { size_t n = (size_t)C * C * 9;
      pack_w_kernel<<<(unsigned)((n + thr - 1) / thr), thr, 0, stream>>>(
          w_c2, g_c2, b_c2, m_c2, v_c2, wc2, bic2, C, C, 9); }

    { size_t n = (size_t)Bn * C * HH * WW;
      f2bf_kernel<<<(unsigned)((n + thr - 1) / thr), thr, 0, stream>>>(x, xbf, n); }

    const unsigned cgrid = (unsigned)((C / 64) * HH * Bn);   // 4096 blocks

    conv_wmma_kernel<256, 9, 0, 0, false><<<cgrid, 256, 0, stream>>>(
        xbf, w12, bi12, nullptr, nullptr, nullptr, p12, nullptr);

    { int n = Bn * D * HH;
      pool_w_kernel<<<(n + thr - 1) / thr, thr, 0, stream>>>(p12, tmp2, Bn, D); }
    { int n = Bn * D * WW;
      pool_h_sum_kernel<<<(n + thr - 1) / thr, thr, 0, stream>>>(p12, tmp2, psum, Bn, D); }

    conv_wmma_kernel<128, 9, 256, 1, false><<<cgrid, 256, 0, stream>>>(
        psum, wpP, biP, xbf, wc1, bic1, rl1, nullptr);

    conv_wmma_kernel<256, 9, 0, 0, true><<<cgrid, 256, 0, stream>>>(
        rl1, wc2, bic2, nullptr, nullptr, nullptr, nullptr, (float*)d_out);
}
